// PagedCausalAttention_34754875359407
// MI455X (gfx1250) — matrix operand order repaired and checked by reading, not compiled, not run on hardware
//
#include <hip/hip_runtime.h>

#define HIDDEN   2048
#define HEADS    16
#define HEAD_DIM 128
#define SEQ      2048
#define BATCH    2
#define M_TOT    (BATCH * SEQ)            // 4096 rows
#define SM_SCALE 0.08838834764831845f     // 1/sqrt(128)
#define NEG_INF_F (-1e30f)

typedef __attribute__((ext_vector_type(16))) _Float16 v16h;
typedef __attribute__((ext_vector_type(8)))  _Float16 v8h;
typedef __attribute__((ext_vector_type(8)))  float    v8f;

__device__ __forceinline__ v16h cat8(v8h lo, v8h hi) {
  return __builtin_shufflevector(lo, hi, 0, 1, 2, 3, 4, 5, 6, 7,
                                         8, 9, 10, 11, 12, 13, 14, 15);
}

// --- gfx1250 async global->LDS copy (ASYNCcnt-tracked, per-lane 16B) -------
__device__ __forceinline__ void async_g2l_b128(_Float16* lds_dst,
                                               const _Float16* gsrc) {
  // Low 32 bits of a generic LDS pointer are the LDS byte offset (aperture).
  unsigned int l = (unsigned int)(unsigned long long)lds_dst;
  asm volatile("global_load_async_to_lds_b128 %0, %1, off"
               :: "v"(l), "v"(gsrc)
               : "memory");
}

__device__ __forceinline__ void wait_asynccnt0() {
#if defined(__has_builtin)
#if __has_builtin(__builtin_amdgcn_s_wait_asynccnt)
  __builtin_amdgcn_s_wait_asynccnt(0);
#else
  asm volatile("s_wait_asynccnt 0" ::: "memory");
#endif
#else
  asm volatile("s_wait_asynccnt 0" ::: "memory");
#endif
}

// ---------------------------------------------------------------------------
// f32 -> f16 staging, 8 elements/thread (2x b128 load, 1x b128 store)
// ---------------------------------------------------------------------------
__global__ void cvt_f32_f16_v8(const float* __restrict__ src,
                               _Float16* __restrict__ dst, int n8) {
  int i = blockIdx.x * blockDim.x + threadIdx.x;
  if (i >= n8) return;
  const float4* s4 = (const float4*)src + 2 * (size_t)i;
  float4 a = s4[0], b = s4[1];
  v8h o = {(_Float16)a.x, (_Float16)a.y, (_Float16)a.z, (_Float16)a.w,
           (_Float16)b.x, (_Float16)b.y, (_Float16)b.z, (_Float16)b.w};
  *(v8h*)(dst + 8 * (size_t)i) = o;
}

// ---------------------------------------------------------------------------
// LDS-tiled f16 GEMM: C (MxN) = A (MxK f16 row-major) * B (KxN f16 row-major).
// 256 threads = 8 waves; block tile 128x128, K-step 32.
// A tile staged via global_load_async_to_lds_b128 (ASYNCcnt path); B tile is
// register-transposed into LDS so both fragments are contiguous ds_load_b128.
// Wave computes a 32x64 subtile = 2x4 WMMA tiles (8 v_wmma per K-step).
// ---------------------------------------------------------------------------
#define BM 128
#define BN 128
#define BK 32
#define LSTR 40   // padded LDS row stride in f16 (80B, keeps 16B alignment)

template <bool F16OUT>
__global__ __launch_bounds__(256)
void gemm_wmma_f16t(const _Float16* __restrict__ A, const _Float16* __restrict__ B,
                    float* __restrict__ Cf, _Float16* __restrict__ Ch,
                    int M, int N, int K) {
  __shared__ __align__(16) _Float16 As[BM][LSTR];
  __shared__ __align__(16) _Float16 Bs[BN][LSTR];   // transposed: [n][k]

  const int tid  = threadIdx.x;
  const int lane = tid & 31;
  const int wave = tid >> 5;
  const int n16  = lane & 15;
  const int kh   = lane >> 4;
  const int wm   = wave & 3;          // 4 row groups of 32
  const int wn   = wave >> 2;         // 2 col groups of 64
  const int row0 = blockIdx.y * BM;
  const int col0 = blockIdx.x * BN;

  // Chunk mappings for cooperative staging (512 chunks of 8 f16 each).
  const int ar0 = tid >> 2, ac0 = (tid & 3) * 8;               // A chunk 0
  const int ar1 = (tid + 256) >> 2, ac1 = ((tid + 256) & 3) * 8;
  const int bk0 = tid & 31, bn0 = (tid >> 5) * 8;              // B chunk 0
  const int bk1 = (tid + 256) & 31, bn1 = ((tid + 256) >> 5) * 8;

  v8f acc[2][4];
#pragma unroll
  for (int t = 0; t < 2; ++t)
#pragma unroll
    for (int u = 0; u < 4; ++u) acc[t][u] = (v8f){};

  for (int kk = 0; kk < K; kk += BK) {
    // --- A tile via async global->LDS (no transpose needed) ---
    async_g2l_b128(&As[ar0][ac0], A + (size_t)(row0 + ar0) * K + kk + ac0);
    async_g2l_b128(&As[ar1][ac1], A + (size_t)(row0 + ar1) * K + kk + ac1);

    // --- B tile (32x128) register-transposed into Bs[n][k] ---
    {
      v8h val = *(const v8h*)(B + (size_t)(kk + bk0) * N + col0 + bn0);
#pragma unroll
      for (int e = 0; e < 8; ++e) Bs[bn0 + e][bk0] = val[e];
      val = *(const v8h*)(B + (size_t)(kk + bk1) * N + col0 + bn1);
#pragma unroll
      for (int e = 0; e < 8; ++e) Bs[bn1 + e][bk1] = val[e];
    }

    // --- prefetch next K-step tiles into L2 while we compute ---
    if (kk + BK < K) {
      __builtin_prefetch(A + (size_t)(row0 + ar0) * K + kk + BK + ac0, 0, 3);
      __builtin_prefetch(B + (size_t)(kk + BK + bk0) * N + col0 + bn0, 0, 3);
    }

    wait_asynccnt0();
    __syncthreads();

    // --- fragments from LDS (each = two 16B ds_load_b128) ---
    v16h afr[2], bfr[4];
#pragma unroll
    for (int t = 0; t < 2; ++t) {
      const _Float16* ap = &As[wm * 32 + t * 16 + n16][0];
      afr[t] = cat8(*(const v8h*)(ap + kh * 8),
                    *(const v8h*)(ap + 16 + kh * 8));
    }
#pragma unroll
    for (int u = 0; u < 4; ++u) {
      const _Float16* bp = &Bs[wn * 64 + u * 16 + n16][kh * 8];
      bfr[u] = cat8(*(const v8h*)(bp), *(const v8h*)(bp + 16));
    }
#pragma unroll
    for (int t = 0; t < 2; ++t)
#pragma unroll
      for (int u = 0; u < 4; ++u)
        acc[t][u] = __builtin_amdgcn_wmma_f32_16x16x32_f16(
            false, afr[t], false, bfr[u], (short)0, acc[t][u], false, false);
    __syncthreads();
  }

  // --- epilogue ---
#pragma unroll
  for (int t = 0; t < 2; ++t) {
#pragma unroll
    for (int r = 0; r < 8; ++r) {
      int row = row0 + wm * 32 + t * 16 + kh * 8 + r;
#pragma unroll
      for (int u = 0; u < 4; ++u) {
        int col = col0 + wn * 64 + u * 16 + n16;
        if (F16OUT) Ch[(size_t)row * N + col] = (_Float16)acc[t][u][r];
        else        Cf[(size_t)row * N + col] = acc[t][u][r];
      }
    }
  }
}

// ---------------------------------------------------------------------------
// RoPE on q,k (f16, in place) + emit f32 k_new/v_new to d_out + build the
// transposed V copy vt[(b,h,d,s)] (f16) used by the P*V stage of attention.
// ---------------------------------------------------------------------------
__global__ void rope_kv(_Float16* __restrict__ q, _Float16* __restrict__ k,
                        const _Float16* __restrict__ v,
                        const int* __restrict__ positions,
                        float* __restrict__ k_out, float* __restrict__ v_out,
                        _Float16* __restrict__ vt) {
  int tid = blockIdx.x * blockDim.x + threadIdx.x;
  int i = tid & 63;
  int h = (tid >> 6) & 15;
  int s = (tid >> 10) & 2047;
  int b = tid >> 21;
  size_t base = (size_t)(b * SEQ + s) * HIDDEN + h * HEAD_DIM;

  float p   = (float)positions[s];
  float ang = p * powf(10000.0f, -(float)i / 64.0f);  // 10000^(-2i/128)
  float c = cosf(ang), sn = sinf(ang);

  float x1 = (float)q[base + i], x2 = (float)q[base + i + 64];
  q[base + i]      = (_Float16)(x1 * c - x2 * sn);
  q[base + i + 64] = (_Float16)(x2 * c + x1 * sn);

  x1 = (float)k[base + i]; x2 = (float)k[base + i + 64];
  float k1 = x1 * c - x2 * sn;
  float k2 = x2 * c + x1 * sn;
  k[base + i]      = (_Float16)k1;
  k[base + i + 64] = (_Float16)k2;
  k_out[base + i]      = k1;
  k_out[base + i + 64] = k2;

  float v1 = (float)v[base + i], v2 = (float)v[base + i + 64];
  v_out[base + i]      = v1;
  v_out[base + i + 64] = v2;
  size_t vtbase = (size_t)(b * HEADS + h) * HEAD_DIM * SEQ + s;
  vt[vtbase + (size_t)i * SEQ]        = (_Float16)v1;
  vt[vtbase + (size_t)(i + 64) * SEQ] = (_Float16)v2;
}

// ---------------------------------------------------------------------------
// Flash attention, one wave32 per (b, h, 16-row Q tile), f16 operands.
// Q/K fragments are contiguous 16B global loads; P*V uses the transposed V
// copy so its B-fragments are contiguous too. Online softmax via LDS.
// ---------------------------------------------------------------------------
__global__ __launch_bounds__(32)
void flash_attn_wmma(const _Float16* __restrict__ q, const _Float16* __restrict__ k,
                     const _Float16* __restrict__ vt, _Float16* __restrict__ attn) {
  const int lane = threadIdx.x;
  const int n16  = lane & 15;
  const int kh   = lane >> 4;
  const int qt   = blockIdx.x;
  const int h    = blockIdx.y;
  const int b    = blockIdx.z;

  __shared__ __align__(16) float    sc[16][32];
  __shared__ __align__(16) _Float16 pbuf[16][32];
  __shared__ float rowstat[16];

  const int qmax = qt * 16 + 15;

  v16h aq[4];
  {
    const _Float16* qrow =
        q + (size_t)(b * SEQ + qt * 16 + n16) * HIDDEN + h * HEAD_DIM;
#pragma unroll
    for (int c = 0; c < 4; ++c)
      aq[c] = cat8(*(const v8h*)(qrow + c * 32 + kh * 8),
                   *(const v8h*)(qrow + c * 32 + 16 + kh * 8));
  }

  v8f o[8] = {};
  float m_run = NEG_INF_F, l_run = 0.0f;   // valid in lanes 0..15 (row = lane)

  const int ktmax = qmax >> 5;
  for (int kt = 0; kt <= ktmax; ++kt) {
    const int jbase = kt * 32;

    // --- scores for 32 keys (two 16x16 tiles) ---
#pragma unroll
    for (int hh = 0; hh < 2; ++hh) {
      const int j0 = jbase + hh * 16;
      v8f s8 = {};
      const _Float16* krow =
          k + (size_t)(b * SEQ + j0 + n16) * HIDDEN + h * HEAD_DIM + kh * 8;
#pragma unroll
      for (int c = 0; c < 4; ++c) {
        v16h bk = cat8(*(const v8h*)(krow + c * 32),
                       *(const v8h*)(krow + c * 32 + 16));
        s8 = __builtin_amdgcn_wmma_f32_16x16x32_f16(false, aq[c], false, bk,
                                                    (short)0, s8, false, false);
      }
      const int kabs = j0 + n16;
#pragma unroll
      for (int r = 0; r < 8; ++r) {
        int qabs = qt * 16 + kh * 8 + r;
        sc[kh * 8 + r][hh * 16 + n16] =
            (kabs <= qabs) ? s8[r] * SM_SCALE : NEG_INF_F;
      }
    }
    __syncthreads();

    // --- online softmax row stats (lane = row) ---
    if (lane < 16) {
      const int row = lane;
      float mt = NEG_INF_F;
#pragma unroll
      for (int j2 = 0; j2 < 32; ++j2) mt = fmaxf(mt, sc[row][j2]);
      float mnew = fmaxf(m_run, mt);
      float corr = expf(m_run - mnew);
      float ssum = 0.0f;
#pragma unroll
      for (int j2 = 0; j2 < 32; ++j2) {
        float pv = expf(sc[row][j2] - mnew);
        ssum += pv;
        pbuf[row][j2] = (_Float16)pv;
      }
      l_run = l_run * corr + ssum;
      m_run = mnew;
      rowstat[row] = corr;
    }
    __syncthreads();

    // --- rescale running output ---
#pragma unroll
    for (int r = 0; r < 8; ++r) {
      float corr = rowstat[kh * 8 + r];
#pragma unroll
      for (int c = 0; c < 8; ++c) o[c][r] *= corr;
    }

    // --- P (16x32 f16 A fragment) x V (32x16 chunks from transposed V) ---
    v16h ap = cat8(*(const v8h*)(&pbuf[n16][kh * 8]),
                   *(const v8h*)(&pbuf[n16][16 + kh * 8]));
#pragma unroll
    for (int c = 0; c < 8; ++c) {
      const _Float16* vrow =
          vt + (size_t)((b * HEADS + h) * HEAD_DIM + c * 16 + n16) * SEQ +
          jbase + kh * 8;
      v16h bv = cat8(*(const v8h*)(vrow), *(const v8h*)(vrow + 16));
      o[c] = __builtin_amdgcn_wmma_f32_16x16x32_f16(false, ap, false, bv,
                                                    (short)0, o[c], false, false);
    }
    __syncthreads();
  }

  // --- finalize ---
  if (lane < 16) rowstat[lane] = l_run;
  __syncthreads();
#pragma unroll
  for (int r = 0; r < 8; ++r) {
    float linv = 1.0f / rowstat[kh * 8 + r];
    _Float16* orow =
        attn + (size_t)(b * SEQ + qt * 16 + kh * 8 + r) * HIDDEN + h * HEAD_DIM;
#pragma unroll
    for (int c = 0; c < 8; ++c)
      orow[c * 16 + n16] = (_Float16)(o[c][r] * linv);
  }
}

// ---------------------------------------------------------------------------
extern "C" void kernel_launch(void* const* d_in, const int* in_sizes, int n_in,
                              void* d_out, int out_size, void* d_ws, size_t ws_size,
                              hipStream_t stream) {
  const float* hs        = (const float*)d_in[0];
  const int*   positions = (const int*)  d_in[1];
  const float* Wq        = (const float*)d_in[2];
  const float* Wk        = (const float*)d_in[3];
  const float* Wv        = (const float*)d_in[4];
  const float* Wo        = (const float*)d_in[5];

  const size_t per   = (size_t)M_TOT * HIDDEN;    // 8,388,608
  const size_t per_w = (size_t)HIDDEN * HIDDEN;   // 4,194,304

  float* out   = (float*)d_out;
  float* k_out = out + per;
  float* v_out = k_out + per;

  _Float16* hs_h   = (_Float16*)d_ws;
  _Float16* wq_h   = hs_h + per;
  _Float16* wk_h   = wq_h + per_w;
  _Float16* wv_h   = wk_h + per_w;
  _Float16* wo_h   = wv_h + per_w;
  _Float16* q_h    = wo_h + per_w;
  _Float16* k_h    = q_h + per;
  _Float16* v_h    = k_h + per;
  _Float16* vt_h   = v_h + per;
  _Float16* attn_h = vt_h + per;

  // --- one-time f16 staging (8 elems/thread) ---
  cvt_f32_f16_v8<<<(int)(per / 8 / 256), 256, 0, stream>>>(hs, hs_h,
                                                           (int)(per / 8));
  cvt_f32_f16_v8<<<(int)(per_w / 8 / 256), 256, 0, stream>>>(Wq, wq_h,
                                                             (int)(per_w / 8));
  cvt_f32_f16_v8<<<(int)(per_w / 8 / 256), 256, 0, stream>>>(Wk, wk_h,
                                                             (int)(per_w / 8));
  cvt_f32_f16_v8<<<(int)(per_w / 8 / 256), 256, 0, stream>>>(Wv, wv_h,
                                                             (int)(per_w / 8));
  cvt_f32_f16_v8<<<(int)(per_w / 8 / 256), 256, 0, stream>>>(Wo, wo_h,
                                                             (int)(per_w / 8));

  dim3 blk(256);
  dim3 gg(HIDDEN / BN, M_TOT / BM);   // (16, 32)

  // --- QKV projections (f16 out) ---
  gemm_wmma_f16t<true><<<gg, blk, 0, stream>>>(hs_h, wq_h, nullptr, q_h,
                                               M_TOT, HIDDEN, HIDDEN);
  gemm_wmma_f16t<true><<<gg, blk, 0, stream>>>(hs_h, wk_h, nullptr, k_h,
                                               M_TOT, HIDDEN, HIDDEN);
  gemm_wmma_f16t<true><<<gg, blk, 0, stream>>>(hs_h, wv_h, nullptr, v_h,
                                               M_TOT, HIDDEN, HIDDEN);

  // --- RoPE + kv outputs + transposed V ---
  int rope_threads = BATCH * SEQ * HEADS * 64;
  rope_kv<<<rope_threads / 256, 256, 0, stream>>>(q_h, k_h, v_h, positions,
                                                  k_out, v_out, vt_h);

  // --- attention ---
  flash_attn_wmma<<<dim3(SEQ / 16, HEADS, BATCH), 32, 0, stream>>>(
      q_h, k_h, vt_h, attn_h);

  // --- output projection (f32 out) ---
  gemm_wmma_f16t<false><<<gg, blk, 0, stream>>>(attn_h, wo_h, out, nullptr,
                                                M_TOT, HIDDEN, HIDDEN);
}
